// GAT_6786048328630
// MI455X (gfx1250) — matrix operand order repaired and checked by reading, not compiled, not run on hardware
//
#include <hip/hip_runtime.h>

typedef __attribute__((ext_vector_type(16))) __bf16 v16bf;
typedef __attribute__((ext_vector_type(8)))  float  v8f;

#define NEG_SLOPE 0.2f

// ---------------------------------------------------------------------------
// Monotone float atomic max: positive floats order like signed ints,
// negative floats order inverted like unsigned ints. Each path is
// non-decreasing in float order, so mixed concurrent use is safe.
// ---------------------------------------------------------------------------
__device__ __forceinline__ void atomicMaxF32(float* addr, float val) {
    if (val >= 0.0f) {
        atomicMax((int*)addr, __float_as_int(val));
    } else {
        atomicMin((unsigned int*)addr, __float_as_uint(val));
    }
}

// ---------------------------------------------------------------------------
// fp32 -> bf16 weight conversion (one-time, tiny)
// ---------------------------------------------------------------------------
__global__ void cvt_bf16_kernel(const float* __restrict__ W,
                                __bf16* __restrict__ Wb, int n) {
    int i = blockIdx.x * blockDim.x + threadIdx.x;
    if (i < n) Wb[i] = (__bf16)W[i];
}

// ---------------------------------------------------------------------------
// Fused H = X(MxK fp32)*Wb(KxF bf16) + attention scores, via WMMA.
// One wave computes a 16-row x F strip: NT = F/16 accumulators, A fragment
// loaded ONCE per K-step and reused across NT WMMAs (X read exactly once).
// Epilogue: each row m lives in one 16-lane half-group (one value per n-tile
// per lane), so ss[m]=h[m,:].a_src and sd[m]=h[m,:].a_dst reduce with 4
// __shfl_xor steps (masks 1/2/4/8 stay inside the half-group).
//
// A frag (16x32 bf16): lane&15 = row M; lane>=16 takes K offsets {8..15,24..31}
// B frag (32x16 bf16): lane&15 = N column; lane>=16 takes K offsets {8..15,24..31}
// C/D   (16x16 f32)  : vgpr r -> M = r + 8*(lane>=16); N = lane&15
// ---------------------------------------------------------------------------
template <int NT>
__global__ __launch_bounds__(256) void gemm_scores_wmma_kernel(
    const float* __restrict__ X, const __bf16* __restrict__ Wb,
    const float* __restrict__ asrc, const float* __restrict__ adst,
    float* __restrict__ H, float* __restrict__ ss, float* __restrict__ sd,
    int M, int K) {
    constexpr int F = NT * 16;
    const int wid   = threadIdx.x >> 5;
    const int lane  = threadIdx.x & 31;
    const int strip = blockIdx.x * 8 + wid;         // 16-row strip index
    if (strip >= (M >> 4)) return;                  // wave-uniform: EXEC all-ones

    const int m0   = strip << 4;
    const int half = lane >> 4;
    const int col  = lane & 15;
    const float* xr = X + (size_t)(m0 + col) * K;

    v8f c[NT];
#pragma unroll
    for (int t = 0; t < NT; ++t) c[t] = v8f{};

    for (int kk = 0; kk < K; kk += 32) {
        // ---- A fragment: 8 floats at kk+8*half, 8 floats at +16 (loaded once) ----
        const int k0 = kk + half * 8;
        const float4 a0 = ((const float4*)(xr + k0))[0];
        const float4 a1 = ((const float4*)(xr + k0))[1];
        const float4 a2 = ((const float4*)(xr + k0 + 16))[0];
        const float4 a3 = ((const float4*)(xr + k0 + 16))[1];
        v16bf A;
        A[0]  = (__bf16)a0.x; A[1]  = (__bf16)a0.y; A[2]  = (__bf16)a0.z; A[3]  = (__bf16)a0.w;
        A[4]  = (__bf16)a1.x; A[5]  = (__bf16)a1.y; A[6]  = (__bf16)a1.z; A[7]  = (__bf16)a1.w;
        A[8]  = (__bf16)a2.x; A[9]  = (__bf16)a2.y; A[10] = (__bf16)a2.z; A[11] = (__bf16)a2.w;
        A[12] = (__bf16)a3.x; A[13] = (__bf16)a3.y; A[14] = (__bf16)a3.z; A[15] = (__bf16)a3.w;
        // ---- reuse A across all NT column tiles ----
        const __bf16* wrow = Wb + (size_t)(kk + half * 8) * F + col;
#pragma unroll
        for (int t = 0; t < NT; ++t) {
            v16bf B;
#pragma unroll
            for (int s = 0; s < 8; ++s) {
                B[s]     = wrow[(size_t)s * F + t * 16];
                B[s + 8] = wrow[(size_t)(s + 16) * F + t * 16];
            }
            c[t] = __builtin_amdgcn_wmma_f32_16x16x32_bf16(false, A, false, B,
                                                           (short)0, c[t], false, false);
        }
    }

    // ---- attention-vector values for this lane's column across all tiles ----
    float av[NT], dv[NT];
#pragma unroll
    for (int t = 0; t < NT; ++t) {
        av[t] = asrc[t * 16 + col];
        dv[t] = adst[t * 16 + col];
    }

    // ---- store H strip + fused per-row score dot products ----
    float* hbase = H + (size_t)(m0 + half * 8) * F + col;
#pragma unroll
    for (int r = 0; r < 8; ++r) {
        float s0 = 0.0f, s1 = 0.0f;
        float* hp = hbase + (size_t)r * F;
#pragma unroll
        for (int t = 0; t < NT; ++t) {
            const float v = c[t][r];
            hp[t * 16] = v;
            s0 += v * av[t];
            s1 += v * dv[t];
        }
        // reduce across the 16-lane half-group (xor 1/2/4/8 stays in-group)
#pragma unroll
        for (int off = 8; off > 0; off >>= 1) {
            s0 += __shfl_xor(s0, off, 32);
            s1 += __shfl_xor(s1, off, 32);
        }
        if (col == 0) {
            const int m = m0 + r + half * 8;
            ss[m] = s0;
            sd[m] = s1;
        }
    }
}

__global__ void init_md_kernel(float* __restrict__ m, float* __restrict__ denom, int N) {
    const int i = blockIdx.x * blockDim.x + threadIdx.x;
    if (i < N) { m[i] = __int_as_float(0xff800000); denom[i] = 0.0f; }
}

// Edge ids [0,E) are real edges; [E,ET) are self-loops (src=dst=e-E).
__global__ void edge_max_kernel(const int* __restrict__ src, const int* __restrict__ dst,
                                const float* __restrict__ ss, const float* __restrict__ sd,
                                float* __restrict__ m, long long E, long long ET) {
    const long long e = blockIdx.x * (long long)blockDim.x + threadIdx.x;
    if (e >= ET) return;
    const int s = (e < E) ? src[e] : (int)(e - E);
    const int d = (e < E) ? dst[e] : (int)(e - E);
    float v = ss[s] + sd[d];
    v = (v > 0.0f) ? v : NEG_SLOPE * v;
    atomicMaxF32(m + d, v);
}

__global__ void edge_sum_kernel(const int* __restrict__ src, const int* __restrict__ dst,
                                const float* __restrict__ ss, const float* __restrict__ sd,
                                const float* __restrict__ m, float* __restrict__ exb,
                                float* __restrict__ denom, long long E, long long ET) {
    const long long e = blockIdx.x * (long long)blockDim.x + threadIdx.x;
    if (e >= ET) return;
    const int s = (e < E) ? src[e] : (int)(e - E);
    const int d = (e < E) ? dst[e] : (int)(e - E);
    float v = ss[s] + sd[d];
    v = (v > 0.0f) ? v : NEG_SLOPE * v;
    const float ex = __expf(v - m[d]);
    exb[e] = ex;
    atomicAdd(denom + d, ex);
}

// ---------------------------------------------------------------------------
// Scatter: agg[dst,:] += (ex/denom[dst]) * h[src,:]
// F/4 lanes cooperate per edge (cshift = log2(F/4)); float4 gathers + fp32 atomics.
// ---------------------------------------------------------------------------
__global__ __launch_bounds__(256) void scatter_kernel(
    const int* __restrict__ src, const int* __restrict__ dst,
    const float* __restrict__ exb, const float* __restrict__ denom,
    const float* __restrict__ H, float* __restrict__ agg,
    long long E, long long ET, int F, int cshift) {
    const int epb = blockDim.x >> cshift;               // edges per block
    const long long e = (long long)blockIdx.x * epb + (threadIdx.x >> cshift);
    if (e >= ET) return;
    const int c = threadIdx.x & ((1 << cshift) - 1);    // float4-chunk within row
    const int s = (e < E) ? src[e] : (int)(e - E);
    const int d = (e < E) ? dst[e] : (int)(e - E);
    const float coef = exb[e] / denom[d];
    const float4 hv = *(const float4*)(H + (size_t)s * F + (c << 2));
    float* ap = agg + (size_t)d * F + (c << 2);
    atomicAdd(ap + 0, coef * hv.x);
    atomicAdd(ap + 1, coef * hv.y);
    atomicAdd(ap + 2, coef * hv.z);
    atomicAdd(ap + 3, coef * hv.w);
}

__global__ void bias_relu_kernel(const float* __restrict__ agg, const float* __restrict__ b,
                                 float* __restrict__ out, long long total, int fmask) {
    const long long i = blockIdx.x * (long long)blockDim.x + threadIdx.x;
    if (i >= total) return;
    const float v = agg[i] + b[(int)(i & fmask)];
    out[i] = (v > 0.0f) ? v : 0.0f;
}

// ---------------------------------------------------------------------------
extern "C" void kernel_launch(void* const* d_in, const int* in_sizes, int n_in,
                              void* d_out, int out_size, void* d_ws, size_t ws_size,
                              hipStream_t stream) {
    const float* x   = (const float*)d_in[0];
    const int*   ei  = (const int*)d_in[1];
    const float* W1  = (const float*)d_in[2];
    const float* as1 = (const float*)d_in[3];
    const float* ad1 = (const float*)d_in[4];
    const float* b1  = (const float*)d_in[5];
    const float* W2  = (const float*)d_in[6];
    const float* as2 = (const float*)d_in[7];
    const float* ad2 = (const float*)d_in[8];
    const float* b2  = (const float*)d_in[9];
    float* out = (float*)d_out;

    const int Fin = 128;
    const int N   = in_sizes[0] / Fin;          // 100000 (divisible by 16)
    const long long E = in_sizes[1] / 2;        // 1600000
    const int F1  = in_sizes[3];                // 128
    const int F2  = in_sizes[7];                // 64
    const int* srcIdx = ei;
    const int* dstIdx = ei + E;
    const long long ET = E + N;
    const size_t NB = (size_t)N;

    // ---- carve workspace (256B aligned chunks) ----
    char* ws = (char*)d_ws;
    size_t off = 0;
    auto carve = [&](size_t bytes) -> void* {
        void* p = ws + off;
        off += (bytes + 255) & ~(size_t)255;
        return p;
    };
    float*  h1   = (float*)carve(NB * F1 * sizeof(float));   // h1; later h2 + agg2
    float*  agg1 = (float*)carve(NB * F1 * sizeof(float));   // agg1 -> x2 in place
    float*  ss   = (float*)carve(NB * sizeof(float));
    float*  sd   = (float*)carve(NB * sizeof(float));
    float*  mbuf = (float*)carve(NB * sizeof(float));
    float*  dnm  = (float*)carve(NB * sizeof(float));
    float*  exb  = (float*)carve((size_t)ET * sizeof(float));
    __bf16* Wb1  = (__bf16*)carve((size_t)Fin * F1 * sizeof(__bf16));
    __bf16* Wb2  = (__bf16*)carve((size_t)F1 * F2 * sizeof(__bf16));
    float*  h2   = h1;                    // N*F2
    float*  agg2 = h1 + NB * (size_t)F2;  // N*F2 (F1 == 2*F2)

    const int TPB = 256;
    // ---- weights to bf16 ----
    cvt_bf16_kernel<<<(Fin * F1 + TPB - 1) / TPB, TPB, 0, stream>>>(W1, Wb1, Fin * F1);
    cvt_bf16_kernel<<<(F1 * F2 + TPB - 1) / TPB, TPB, 0, stream>>>(W2, Wb2, F1 * F2);

    const unsigned edgeBlocks = (unsigned)((ET + TPB - 1) / TPB);
    const unsigned stripBlocks = (unsigned)((N / 16 + 7) / 8);

    // ================= layer 1 (F1 = 128 -> NT = 8) =================
    {
        gemm_scores_wmma_kernel<8><<<stripBlocks, 256, 0, stream>>>(
            x, Wb1, as1, ad1, h1, ss, sd, N, Fin);
        init_md_kernel<<<(N + TPB - 1) / TPB, TPB, 0, stream>>>(mbuf, dnm, N);
        hipMemsetAsync(agg1, 0, NB * F1 * sizeof(float), stream);
        edge_max_kernel<<<edgeBlocks, TPB, 0, stream>>>(srcIdx, dstIdx, ss, sd, mbuf, E, ET);
        edge_sum_kernel<<<edgeBlocks, TPB, 0, stream>>>(srcIdx, dstIdx, ss, sd, mbuf, exb, dnm, E, ET);
        const int cs = 5, epb = 256 >> 5;  // 32 lanes per edge (F1=128)
        scatter_kernel<<<(unsigned)((ET + epb - 1) / epb), 256, 0, stream>>>(
            srcIdx, dstIdx, exb, dnm, h1, agg1, E, ET, F1, cs);
        const long long tot = (long long)NB * F1;
        bias_relu_kernel<<<(unsigned)((tot + TPB - 1) / TPB), TPB, 0, stream>>>(
            agg1, b1, agg1, tot, F1 - 1);   // in place: agg1 becomes x2
    }

    // ================= layer 2 (F2 = 64 -> NT = 4) =================
    {
        gemm_scores_wmma_kernel<4><<<stripBlocks, 256, 0, stream>>>(
            agg1, Wb2, as2, ad2, h2, ss, sd, N, F1);
        init_md_kernel<<<(N + TPB - 1) / TPB, TPB, 0, stream>>>(mbuf, dnm, N);
        hipMemsetAsync(agg2, 0, NB * F2 * sizeof(float), stream);
        edge_max_kernel<<<edgeBlocks, TPB, 0, stream>>>(srcIdx, dstIdx, ss, sd, mbuf, E, ET);
        edge_sum_kernel<<<edgeBlocks, TPB, 0, stream>>>(srcIdx, dstIdx, ss, sd, mbuf, exb, dnm, E, ET);
        const int cs = 4, epb = 256 >> 4;  // 16 lanes per edge (F2=64)
        scatter_kernel<<<(unsigned)((ET + epb - 1) / epb), 256, 0, stream>>>(
            srcIdx, dstIdx, exb, dnm, h2, agg2, E, ET, F2, cs);
        const long long tot = (long long)NB * F2;
        bias_relu_kernel<<<(unsigned)((tot + TPB - 1) / TPB), TPB, 0, stream>>>(
            agg2, b2, out, tot, F2 - 1);
    }
}
